// CriticOld_84456236908765
// MI455X (gfx1250) — compile-verified, matrix operand order checked by reading, not run on hardware
//
#include <hip/hip_runtime.h>

// ---------------- problem constants ----------------
#define NB    30      // nodes per sample
#define KNN   15      // neighbors
#define HID   128
#define EMB   64
#define FEAT  194     // HID + EMB + 2
#define K1    388     // 2*FEAT
#define K1PAD 416     // 13 * 32
#define KT1   13      // K-tiles for layer 1 (K=416)
#define KT2   4       // K-tiles for K=128 GEMMs
#define NT    8       // N-tiles for N=128

typedef __attribute__((ext_vector_type(16))) _Float16 v16h;
typedef __attribute__((ext_vector_type(8)))  float    v8f;

// ---- WMMA 16-bit operand register layout (ISA 7.12.2) ----
// lanes 0-15 hold K {0..7, 16..23}, lanes 16-31 hold K {8..15, 24..31};
// within a lane, half h: K 0..7 -> h = K&7 ; K 16..23(/24..31) -> h = (K&7)+8.
// K offset of half-pair v (0..7):
__device__ __forceinline__ int wmma_kbase(int v, int hi) {
  return (v < 4) ? (hi ? 8 + 2 * v : 2 * v)
                 : (hi ? 24 + 2 * (v - 4) : 16 + 2 * (v - 4));
}

// Inverse map: element (m, k) of a 16xK f16 A matrix -> flat position in a
// swizzled buffer laid out as [kt][lane][16 halves] (512 f16 per 16x32 tile).
__device__ __forceinline__ int a_swz(int m, int k) {
  int kt = k >> 5;
  int kl = k & 31;
  int lane = m + (((kl >> 3) & 1) << 4);
  int h = (kl & 7) | ((kl >> 4) << 3);
  return (kt << 9) + (lane << 4) + h;
}

// A operand: per-lane 32 contiguous bytes -> two ds_load_b128, no repack.
__device__ __forceinline__ v16h load_a_sw(const _Float16* base, int kt, int lane) {
  return *(const v16h*)(base + (kt << 9) + (lane << 4));
}

// B operand: pre-packed global tile, per-lane 32 contiguous bytes.
__device__ __forceinline__ v16h load_b_sw(const _Float16* base, int tile, int lane) {
  return *(const v16h*)(base + ((size_t)tile << 9) + (lane << 4));
}

__device__ __forceinline__ v8f wmma_f16(v16h a, v16h b, v8f c) {
  return __builtin_amdgcn_wmma_f32_16x16x32_f16(false, a, false, b,
                                                (short)0, c, false, false);
}

// ---------------- weight packing: W (N=128 x Ksrc, row-major [n][k]) ->
// B-operand tiles (kt,nt): 512 f16 each in register order -------------
__global__ void pack_b_tiles(const float* __restrict__ W, _Float16* __restrict__ dst,
                             int Ksrc, int KT) {
  int t = blockIdx.x * blockDim.x + threadIdx.x;
  int total = KT * NT * 512;
  if (t >= total) return;
  int tile = t >> 9;
  int r    = t & 511;
  int lane = r >> 4;
  int h    = r & 15;
  int kt = tile / NT, nt = tile % NT;
  int hi = lane >> 4;
  int nl = lane & 15;
  int kl = wmma_kbase(h >> 1, hi) + (h & 1);
  int kg = kt * 32 + kl;
  int ng = nt * 16 + nl;
  float val = (kg < Ksrc) ? W[ng * Ksrc + kg] : 0.0f;
  dst[t] = (_Float16)val;
}

// ---------------- fused critic kernel: one block = one (sample, Q) ----------
__global__ __launch_bounds__(256)
void critic_fused(const float* __restrict__ state, const float* __restrict__ action,
                  const float* __restrict__ Wi1, const float* __restrict__ bi1,
                  const float* __restrict__ emb1,
                  const float* __restrict__ ba,  const float* __restrict__ bb,
                  const float* __restrict__ bc1a, const float* __restrict__ Wc1b,
                  const float* __restrict__ bc1b,
                  const float* __restrict__ Wi2, const float* __restrict__ bi2,
                  const float* __restrict__ emb2,
                  const float* __restrict__ bc2a, const float* __restrict__ Wc2b,
                  const float* __restrict__ bc2b,
                  const _Float16* __restrict__ pWa, const _Float16* __restrict__ pWb,
                  const _Float16* __restrict__ pWc1, const _Float16* __restrict__ pWc2,
                  float* __restrict__ out) {
  const int b   = blockIdx.x;
  const int qi  = blockIdx.y;          // 0 -> Q1, 1 -> Q2
  const int tid = threadIdx.x;
  const int lane = tid & 31;
  const int wave = tid >> 5;           // 0..7 == N-tile index
  const int hi = lane >> 4;
  const int nl = lane & 15;

  const float* Wi  = qi ? Wi2  : Wi1;
  const float* bi  = qi ? bi2  : bi1;
  const float* emb = qi ? emb2 : emb1;
  const float* bca = qi ? bc2a : bc1a;
  const float* Wcb = qi ? Wc2b : Wc1b;
  const float* bcb = qi ? bc2b : bc1b;
  const _Float16* pWc = qi ? pWc2 : pWc1;

  __shared__ __align__(16) float    nodesLds[NB][4];
  __shared__ __align__(16) int      idxLds[NB][KNN];
  __shared__ __align__(16) _Float16 featLds[NB][208];      // FEAT padded to 208
  __shared__ __align__(32) _Float16 AtileSw[KT1 * 512];    // edge A-tile, swizzled
  __shared__ __align__(32) _Float16 h1Sw[KT2 * 512];       // layer-1 out, swizzled
  __shared__ __align__(32) _Float16 nodeHSw[2 * KT2 * 512];// node feats, swizzled
  __shared__ __align__(16) _Float16 gbuf[32 * 136];        // classifier hidden

  // ---- stage 0: build node coordinates (two different layouts) ----
  if (tid < NB * 4) {
    int i = tid >> 2, d = tid & 3;
    float v;
    if (qi == 0) {
      v = (d < 2) ? state[b * 60 + 2 * i + d] : action[b * 60 + 2 * i + (d - 2)];
    } else {
      int x = 4 * i + d;   // cat along dim1 then view(-1,4)  (torch-bug layout)
      v = (x < 60) ? state[b * 60 + x] : action[b * 60 + (x - 60)];
    }
    nodesLds[i][d] = v;
  }
  __syncthreads();

  // ---- stage 1: per-sample kNN (k smallest sq-dists, self excluded) ----
  if (tid < NB) {
    float x0 = nodesLds[tid][0], x1 = nodesLds[tid][1];
    float x2 = nodesLds[tid][2], x3 = nodesLds[tid][3];
    float d[NB];
    bool used[NB];
#pragma unroll
    for (int j = 0; j < NB; ++j) {
      float d0 = x0 - nodesLds[j][0], d1 = x1 - nodesLds[j][1];
      float d2 = x2 - nodesLds[j][2], d3 = x3 - nodesLds[j][3];
      d[j] = d0 * d0 + d1 * d1 + d2 * d2 + d3 * d3;
      used[j] = false;
    }
    for (int r = 0; r < KNN; ++r) {
      float best = 3.0e38f; int bj = 0;
      for (int j = 0; j < NB; ++j)
        if (j != tid && !used[j] && d[j] < best) { best = d[j]; bj = j; }
      used[bj] = true;
      idxLds[tid][r] = bj;
    }
  }

  // ---- stage 2: init features [relu(x@Wi^T+bi) | relu(emb[cat]) | 0,0] ----
  for (int t = tid; t < NB * HID; t += 256) {
    int i = t >> 7, c = t & 127;
    float s = bi[c];
#pragma unroll
    for (int d = 0; d < 4; ++d) s += nodesLds[i][d] * Wi[c * 4 + d];
    featLds[i][c] = (_Float16)fmaxf(s, 0.0f);
  }
  for (int t = tid; t < NB * EMB; t += 256) {
    int i = t / EMB, c = t % EMB;
    featLds[i][HID + c] = (_Float16)fmaxf(emb[(i / 10) * EMB + c], 0.0f);
  }
  for (int t = tid; t < NB * 16; t += 256) {        // tar + pad columns = 0
    int i = t >> 4, c = t & 15;
    featLds[i][192 + c] = (_Float16)0.0f;
  }
  for (int t = tid; t < 2 * HID; t += 256) {        // zero pad rows 30,31
    int m = 14 + (t >> 7);                          // rows 14,15 of M-tile 1
    int k = t & 127;
    nodeHSw[KT2 * 512 + a_swz(m, k)] = (_Float16)0.0f;
  }
  __syncthreads();

  // ---- stage 3: EdgeConv, one node per 16-row tile (15 edges + pad row) ----
  for (int node = 0; node < NB; ++node) {
    // assemble A-tile directly in WMMA register order:
    // row r = edge r of node; cols [x_i | x_j - x_i | 0-pad]
    for (int t = tid; t < 16 * K1PAD; t += 256) {
      int r = t / K1PAD, c = t - r * K1PAD;
      _Float16 val = (_Float16)0.0f;
      if (r < KNN && c < K1) {
        if (c < FEAT) {
          val = featLds[node][c];
        } else {
          int j = idxLds[node][r];
          int cc = c - FEAT;
          val = featLds[j][cc] - featLds[node][cc];
        }
      }
      AtileSw[a_swz(r, c)] = val;
    }
    __syncthreads();

    // layer 1: (16 x 416) @ (416 x 128): each wave owns one 16-col N-tile
    v8f acc = {};
#pragma unroll
    for (int kt = 0; kt < KT1; ++kt) {
      v16h a = load_a_sw(AtileSw, kt, lane);
      v16h bm = load_b_sw(pWa, kt * NT + wave, lane);
      acc = wmma_f16(a, bm, acc);
    }
    {
      const int col = wave * 16 + nl;
      float bav = ba[col];
#pragma unroll
      for (int r = 0; r < 8; ++r) {
        int row = r + 8 * hi;
        h1Sw[a_swz(row, col)] = (_Float16)fmaxf(acc[r] + bav, 0.0f);
      }
    }
    __syncthreads();

    // layer 2: (16 x 128) @ (128 x 128), then bias + max over the 15 edges
    v8f acc2 = {};
#pragma unroll
    for (int kt = 0; kt < KT2; ++kt) {
      v16h a = load_a_sw(h1Sw, kt, lane);
      v16h bm = load_b_sw(pWb, kt * NT + wave, lane);
      acc2 = wmma_f16(a, bm, acc2);
    }
    {
      const int col = wave * 16 + nl;
      float bbv = bb[col];
      float mx = -3.0e38f;
#pragma unroll
      for (int r = 0; r < 8; ++r) {
        if (!(hi && r == 7))                       // row 15 is padding
          mx = fmaxf(mx, acc2[r] + bbv);
      }
      mx = fmaxf(mx, __shfl_xor(mx, 16, 32));      // combine rows 0-7 / 8-14
      if (hi == 0) {
        int mt = node >> 4, m = node & 15;
        nodeHSw[mt * (KT2 * 512) + a_swz(m, col)] =
            (_Float16)fmaxf(mx, 0.0f);             // outer ReLU
      }
    }
    __syncthreads();
  }

  // ---- stage 4: classifier GEMM relu(H @ Wca^T + bca), 32x128x128 ----
#pragma unroll
  for (int mt = 0; mt < 2; ++mt) {
    v8f acc = {};
#pragma unroll
    for (int kt = 0; kt < KT2; ++kt) {
      v16h a = load_a_sw(nodeHSw + mt * (KT2 * 512), kt, lane);
      v16h bm = load_b_sw(pWc, kt * NT + wave, lane);
      acc = wmma_f16(a, bm, acc);
    }
    const int col = wave * 16 + nl;
    float bcv = bca[col];
#pragma unroll
    for (int r = 0; r < 8; ++r) {
      int row = mt * 16 + r + 8 * hi;
      gbuf[row * 136 + col] = (_Float16)fmaxf(acc[r] + bcv, 0.0f);
    }
  }
  __syncthreads();

  // ---- stage 5: scalar head: g @ Wcb^T + bcb ----
  if (tid < NB) {
    float s = bcb[0];
#pragma unroll 8
    for (int c = 0; c < HID; ++c)
      s += (float)gbuf[tid * 136 + c] * Wcb[c];
    out[qi * (512 * NB) + b * NB + tid] = s;
  }
}

// ---------------- host-side launch ----------------
extern "C" void kernel_launch(void* const* d_in, const int* in_sizes, int n_in,
                              void* d_out, int out_size, void* d_ws, size_t ws_size,
                              hipStream_t stream) {
  (void)in_sizes; (void)n_in; (void)out_size; (void)ws_size;
  const float* state   = (const float*)d_in[0];
  const float* action  = (const float*)d_in[1];
  const float* W_init1 = (const float*)d_in[2];
  const float* b_init1 = (const float*)d_in[3];
  const float* emb1    = (const float*)d_in[4];
  const float* W_m1a   = (const float*)d_in[5];
  const float* b_m1a   = (const float*)d_in[6];
  const float* W_m1b   = (const float*)d_in[7];
  const float* b_m1b   = (const float*)d_in[8];
  const float* W_c1a   = (const float*)d_in[9];
  const float* b_c1a   = (const float*)d_in[10];
  const float* W_c1b   = (const float*)d_in[11];
  const float* b_c1b   = (const float*)d_in[12];
  const float* W_init2 = (const float*)d_in[13];
  const float* b_init2 = (const float*)d_in[14];
  const float* emb2    = (const float*)d_in[15];
  const float* W_c2a   = (const float*)d_in[16];
  const float* b_c2a   = (const float*)d_in[17];
  const float* W_c2b   = (const float*)d_in[18];
  const float* b_c2b   = (const float*)d_in[19];

  _Float16* ws   = (_Float16*)d_ws;
  _Float16* pWa  = ws;                       // 13*8*512 = 53248 f16
  _Float16* pWb  = pWa + KT1 * NT * 512;     // 4*8*512  = 16384 f16
  _Float16* pWc1 = pWb + KT2 * NT * 512;
  _Float16* pWc2 = pWc1 + KT2 * NT * 512;

  {
    int t1 = KT1 * NT * 512;
    int t2 = KT2 * NT * 512;
    pack_b_tiles<<<(t1 + 255) / 256, 256, 0, stream>>>(W_m1a, pWa, K1, KT1);
    pack_b_tiles<<<(t2 + 255) / 256, 256, 0, stream>>>(W_m1b, pWb, HID, KT2);
    pack_b_tiles<<<(t2 + 255) / 256, 256, 0, stream>>>(W_c1a, pWc1, HID, KT2);
    pack_b_tiles<<<(t2 + 255) / 256, 256, 0, stream>>>(W_c2a, pWc2, HID, KT2);
  }

  dim3 grid(512, 2, 1);
  critic_fused<<<grid, 256, 0, stream>>>(
      state, action,
      W_init1, b_init1, emb1,
      b_m1a, b_m1b,
      b_c1a, W_c1b, b_c1b,
      W_init2, b_init2, emb2,
      b_c2a, W_c2b, b_c2b,
      pWa, pWb, pWc1, pWc2,
      (float*)d_out);
}
